// GFAEncoder_64604898066866
// MI455X (gfx1250) — matrix operand order repaired and checked by reading, not compiled, not run on hardware
//
#include <hip/hip_runtime.h>
#include <hip/hip_bf16.h>
#include <math.h>

// ---------- types ----------
typedef unsigned short u16;
typedef u16   v8u   __attribute__((ext_vector_type(8)));
typedef u16   v16u  __attribute__((ext_vector_type(16)));
typedef __bf16 v16bf __attribute__((ext_vector_type(16)));
typedef float v8f   __attribute__((ext_vector_type(8)));

union BF16x16 { v16u u; v16bf b; };

__device__ __forceinline__ u16 f2bf(float f) {
    unsigned u = __float_as_uint(f);
    unsigned r = u + 0x7fffu + ((u >> 16) & 1u);   // round to nearest even
    return (u16)(r >> 16);
}

// ---------- CDNA5 async global->LDS copy (ASYNCcnt-tracked, 512B per wave-instr) ----------
__device__ __forceinline__ void async_g2l_b128(unsigned lds_off, const u16* g) {
    asm volatile("global_load_async_to_lds_b128 %0, %1, off"
                 :: "v"(lds_off), "v"(g) : "memory");
}
__device__ __forceinline__ void wait_async0() {
    asm volatile("s_wait_asynccnt 0" ::: "memory");
}

// ---------- problem constants ----------
// B=16, C=256, H=W=56, Hp=Wp=28, N=784, heads=8, dh=32, E=4
#define Bb   16
#define Cc   256
#define Hh   56
#define Ww   56
#define Nt   784      // 28*28 tokens per image
#define M1   12544    // B*Nt
#define M2   50176    // B*H*W
#define NH   8
#define DH   32

// ---------- generic bf16 WMMA GEMM:  out[M,N] = A[M,K] * W[N,K]^T (+ epilogue) ----------
// grid = (N/128, M/64), block = 256 (8 waves). Each wave: 16(M) x 64(N).
// A/B tiles double-buffered in LDS via global_load_async_to_lds_b128.
__global__ void gfa_gemm_bf16(const u16* __restrict__ A, const u16* __restrict__ Bt,
                              const float* __restrict__ bias, const float* __restrict__ res,
                              float* __restrict__ outF, u16* __restrict__ outB,
                              int M, int N, int K, int act, int out_mode, int ct_p) {
    __shared__ u16 aBuf[2][64 * 32];    // 4 KB per buffer
    __shared__ u16 bBuf[2][128 * 32];   // 8 KB per buffer

    int tid  = threadIdx.x;
    int wave = tid >> 5, lane = tid & 31;
    int wm = wave >> 1, wn = wave & 1;
    int mblock = blockIdx.y * 64;
    int nblock = blockIdx.x * 128;
    int l15    = lane & 15;
    int khalfA = (lane >> 4) * 16;   // A frag: elems 0..15 -> K ka..ka+15 (contiguous)
    int khalfB = (lane >> 4) * 16;   // B frag: elems 0..15 -> K kb..kb+15 (contiguous)

    // per-thread async-copy chunk assignment (16B granules)
    int ar = tid >> 2, aq = tid & 3;                 // A: 256 chunks (64 rows x 4)
    const u16* agsrc = A + (size_t)(mblock + ar) * K + aq * 8;
    unsigned   aldst0 = (unsigned)(size_t)&aBuf[0][ar * 32 + aq * 8];
    unsigned   aldst1 = (unsigned)(size_t)&aBuf[1][ar * 32 + aq * 8];
    int br0 = tid >> 2, bq0 = tid & 3;               // B: 512 chunks (128 rows x 4), 2 per thread
    int br1 = (tid + 256) >> 2, bq1 = tid & 3;
    const u16* bgsrc0 = Bt + (size_t)(nblock + br0) * K + bq0 * 8;
    const u16* bgsrc1 = Bt + (size_t)(nblock + br1) * K + bq1 * 8;
    unsigned   bldst0[2] = { (unsigned)(size_t)&bBuf[0][br0 * 32 + bq0 * 8],
                             (unsigned)(size_t)&bBuf[1][br0 * 32 + bq0 * 8] };
    unsigned   bldst1[2] = { (unsigned)(size_t)&bBuf[0][br1 * 32 + bq1 * 8],
                             (unsigned)(size_t)&bBuf[1][br1 * 32 + bq1 * 8] };

    v8f acc[4];
    acc[0] = v8f{}; acc[1] = v8f{}; acc[2] = v8f{}; acc[3] = v8f{};

    int nsteps = K >> 5;
    // prologue: stage K-step 0 into buffer 0
    async_g2l_b128(aldst0, agsrc);
    async_g2l_b128(bldst0[0], bgsrc0);
    async_g2l_b128(bldst1[0], bgsrc1);

    for (int s = 0; s < nsteps; ++s) {
        int cur = s & 1;
        wait_async0();
        __syncthreads();
        if (s + 1 < nsteps) {
            int nxt = cur ^ 1;
            size_t ko = (size_t)(s + 1) << 5;
            async_g2l_b128(cur ? aldst0 : aldst1, agsrc + ko);   // nxt buffer
            async_g2l_b128(bldst0[nxt], bgsrc0 + ko);
            async_g2l_b128(bldst1[nxt], bgsrc1 + ko);
        }
        // fragments from LDS
        const u16* aL = &aBuf[cur][(wm * 16 + l15) * 32 + khalfA];
        v8u alo = *(const v8u*)(aL);
        v8u ahi = *(const v8u*)(aL + 8);
        BF16x16 af;
        af.u = __builtin_shufflevector(alo, ahi, 0,1,2,3,4,5,6,7,8,9,10,11,12,13,14,15);

        BF16x16 b0, b1, b2, b3;
        b0.u = *(const v16u*)&bBuf[cur][(wn * 64 +  0 + l15) * 32 + khalfB];
        b1.u = *(const v16u*)&bBuf[cur][(wn * 64 + 16 + l15) * 32 + khalfB];
        b2.u = *(const v16u*)&bBuf[cur][(wn * 64 + 32 + l15) * 32 + khalfB];
        b3.u = *(const v16u*)&bBuf[cur][(wn * 64 + 48 + l15) * 32 + khalfB];

        acc[0] = __builtin_amdgcn_wmma_f32_16x16x32_bf16(false, af.b, false, b0.b, (short)0, acc[0], false, false);
        acc[1] = __builtin_amdgcn_wmma_f32_16x16x32_bf16(false, af.b, false, b1.b, (short)0, acc[1], false, false);
        acc[2] = __builtin_amdgcn_wmma_f32_16x16x32_bf16(false, af.b, false, b2.b, (short)0, acc[2], false, false);
        acc[3] = __builtin_amdgcn_wmma_f32_16x16x32_bf16(false, af.b, false, b3.b, (short)0, acc[3], false, false);
    }

    // C/D layout: lane<16: M=r, N=lane ; lane>=16: M=8+r, N=lane-16
    int mbase = mblock + wm * 16;
    int nbase = nblock + wn * 64;
    int rowoff = mbase + ((lane >> 4) << 3);
#pragma unroll
    for (int sub = 0; sub < 4; ++sub) {
        int col = nbase + sub * 16 + l15;
        float bcol = bias[col];
#pragma unroll
        for (int r = 0; r < 8; ++r) {
            int row = rowoff + r;
            float v = acc[sub][r] + bcol;
            if (act) v = 0.5f * v * (1.0f + erff(v * 0.70710678118654752f));
            if (out_mode == 0) {
                size_t o = (size_t)row * N + col;
                if (res)  v += res[o];
                if (outF) outF[o] = v;
                if (outB) outB[o] = f2bf(v);
            } else if (out_mode == 1) {
                // transposed-conv scatter: row = b*784 + i*28 + j -> NHWC (b, 2i+di, 2j+dj, col)
                int b  = row / Nt, rem = row - b * Nt;
                int ii = rem / 28, jj = rem - ii * 28;
                int di = ct_p >> 1, dj = ct_p & 1;
                size_t o = (((size_t)b * Hh + (2 * ii + di)) * Ww + (2 * jj + dj)) * Cc + col;
                outF[o] = v;
            } else {
                // final: row = b*3136 + hh*56 + ww ; write NCHW + inp0 residual
                int b  = row / 3136, rem = row - b * 3136;
                int hh = rem / Ww, ww2 = rem - hh * Ww;
                size_t o = (((size_t)b * Cc + col) * Hh + hh) * Ww + ww2;
                outF[o] = res[o] + v;
            }
        }
    }
}

// ---------- weight conversion ----------
__global__ void gfa_f2bf(const float* __restrict__ in, u16* __restrict__ out, int n) {
    int i = blockIdx.x * 256 + threadIdx.x;
    if (i < n) out[i] = f2bf(in[i]);
}

// pack ct_w[k][n][di][dj] -> wct[(p*256+n)*256+k], p = di*2+dj
__global__ void gfa_ctpack(const float* __restrict__ w, u16* __restrict__ out) {
    int i = blockIdx.x * 256 + threadIdx.x;
    if (i < 4 * 256 * 256) {
        int p = i >> 16, nn = (i >> 8) & 255, kk = i & 255;
        int di = p >> 1, dj = p & 1;
        out[i] = f2bf(w[(((size_t)kk * 256 + nn) * 2 + di) * 2 + dj]);
    }
}

// ---------- depthwise 3x3 conv + residual + 2x2 avg pool ----------
// grid = B*C, block = 256 ; pooled[b*C+c][n] layout [B,C,784]
__global__ void gfa_dwpool(const float* __restrict__ x, const float* __restrict__ dww,
                           const float* __restrict__ dwb, float* __restrict__ pooled) {
    __shared__ float sm[Hh * Ww];
    int bc = blockIdx.x;
    int c  = bc & (Cc - 1);
    const float* xp = x + (size_t)bc * (Hh * Ww);
    for (int i = threadIdx.x; i < Hh * Ww; i += 256) sm[i] = xp[i];
    float wv[9];
#pragma unroll
    for (int i = 0; i < 9; ++i) wv[i] = dww[c * 9 + i];
    float bb = dwb[c];
    __syncthreads();
    for (int o = threadIdx.x; o < Nt; o += 256) {
        int oi = o / 28, oj = o - oi * 28;
        float s = 0.0f;
#pragma unroll
        for (int u = 0; u < 2; ++u)
#pragma unroll
            for (int vv = 0; vv < 2; ++vv) {
                int pi = 2 * oi + u, pj = 2 * oj + vv;
                float a = bb;
#pragma unroll
                for (int ky = 0; ky < 3; ++ky)
#pragma unroll
                    for (int kx = 0; kx < 3; ++kx) {
                        int yy = pi + ky - 1, xx = pj + kx - 1;
                        if (yy >= 0 && yy < Hh && xx >= 0 && xx < Ww)
                            a += sm[yy * Ww + xx] * wv[ky * 3 + kx];
                    }
                s += a + sm[pi * Ww + pj];     // conv + residual
            }
        pooled[(size_t)bc * Nt + o] = s * 0.25f;
    }
}

// ---------- block reduction helper ----------
__device__ __forceinline__ float gfa_breduce(float v, float* sm) {
    int tid = threadIdx.x;
    sm[tid] = v; __syncthreads();
    for (int s = 128; s > 0; s >>= 1) {
        if (tid < s) sm[tid] += sm[tid + s];
        __syncthreads();
    }
    float r = sm[0]; __syncthreads();
    return r;
}

// ---------- LN over C for pooled tokens (channel-major input) ----------
__global__ void gfa_ln1(const float* __restrict__ pooled, const float* __restrict__ g,
                        const float* __restrict__ bt, float* __restrict__ tf, u16* __restrict__ tb) {
    __shared__ float red[256];
    int token = blockIdx.x;
    int b = token / Nt, n = token - b * Nt;
    int c = threadIdx.x;
    float v = pooled[((size_t)(b * Cc + c)) * Nt + n];
    float mean = gfa_breduce(v, red) * (1.0f / Cc);
    float d = v - mean;
    float var = gfa_breduce(d * d, red) * (1.0f / Cc);
    float o = d * rsqrtf(var + 1e-6f) * g[c] + bt[c];
    size_t idx = (size_t)token * Cc + c;
    tf[idx] = o;
    tb[idx] = f2bf(o);
}

// ---------- LN over C, row-major input, bf16 out ----------
__global__ void gfa_ln2(const float* __restrict__ in, const float* __restrict__ g,
                        const float* __restrict__ bt, u16* __restrict__ outb) {
    __shared__ float red[256];
    int token = blockIdx.x;
    int c = threadIdx.x;
    size_t idx = (size_t)token * Cc + c;
    float v = in[idx];
    float mean = gfa_breduce(v, red) * (1.0f / Cc);
    float d = v - mean;
    float var = gfa_breduce(d * d, red) * (1.0f / Cc);
    outb[idx] = f2bf(d * rsqrtf(var + 1e-6f) * g[c] + bt[c]);
}

// ---------- q/k L2 norms over tokens: rq/rk[(b*8+h)*32+d] = 1/max(||.||,1e-12) ----------
__global__ void gfa_norms(const float* __restrict__ qkv, float* __restrict__ rq, float* __restrict__ rk) {
    __shared__ float red[256];
    int bh = blockIdx.x; int b = bh >> 3, h = bh & 7;
    int tid = threadIdx.x, g = tid >> 2, tl = tid & 3;
    int qk = g >> 5, d = g & 31;
    const float* base = qkv + (size_t)b * Nt * 768 + (size_t)qk * 256 + h * DH + d;
    float s = 0.0f;
    for (int n = tl; n < Nt; n += 4) { float v = base[(size_t)n * 768]; s += v * v; }
    red[tid] = s; __syncthreads();
    if (tl == 0) {
        float t = red[tid] + red[tid + 1] + red[tid + 2] + red[tid + 3];
        float sc = 1.0f / fmaxf(sqrtf(t), 1e-12f);
        (qk ? rk : rq)[bh * DH + d] = sc;
    }
}

// ---------- channel cross-covariance attention + mask + softmax ----------
// one block per (b,h); q,k tiled through LDS in 7 chunks of 112 tokens
__global__ void gfa_attn(const float* __restrict__ qkv, const float* __restrict__ rq,
                         const float* __restrict__ rk, const float* __restrict__ mask_u,
                         const float* __restrict__ temp, float* __restrict__ attn_out) {
    __shared__ float qs[32 * 112];
    __shared__ float ks2[32 * 112];
    __shared__ float at[32 * 33];
    int bh = blockIdx.x; int b = bh >> 3, h = bh & 7;
    int tid = threadIdx.x;
    float acc[4] = {0.f, 0.f, 0.f, 0.f};
    const float* base = qkv + (size_t)b * Nt * 768 + h * DH;
    for (int t7 = 0; t7 < 7; ++t7) {
        int nb = t7 * 112;
        for (int idx = tid; idx < 32 * 112; idx += 256) {
            int d = idx / 112, nn = idx - d * 112;
            size_t off = (size_t)(nb + nn) * 768 + d;
            qs[idx]  = base[off];
            ks2[idx] = base[off + 256];
        }
        __syncthreads();
#pragma unroll
        for (int pp = 0; pp < 4; ++pp) {
            int p = pp * 256 + tid; int d = p >> 5, e = p & 31;
            const float* qd = qs  + d * 112;
            const float* ke = ks2 + e * 112;
            float s = 0.0f;
            for (int nn = 0; nn < 112; ++nn) s += qd[nn] * ke[nn];
            acc[pp] += s;
        }
        __syncthreads();
    }
    float tmph = temp[h];
#pragma unroll
    for (int pp = 0; pp < 4; ++pp) {
        int p = pp * 256 + tid; int d = p >> 5, e = p & 31;
        float s = acc[pp] * rq[bh * DH + d] * rk[bh * DH + e] * tmph;
        if (mask_u[((size_t)bh * DH + d) * DH + e] < 0.2f) s -= 1e12f;
        at[d * 33 + e] = s;
    }
    __syncthreads();
    if (tid < 32) {
        int d = tid;
        float m = -3.0e38f;
        for (int e = 0; e < 32; ++e) m = fmaxf(m, at[d * 33 + e]);
        float ssum = 0.0f;
        for (int e = 0; e < 32; ++e) { float x = expf(at[d * 33 + e] - m); at[d * 33 + e] = x; ssum += x; }
        float inv = 1.0f / ssum;
        for (int e = 0; e < 32; ++e)
            attn_out[(size_t)bh * 1024 + d * 32 + e] = at[d * 33 + e] * inv;
    }
}

// ---------- out = attn @ v, written token-major bf16 [M1, C] ----------
__global__ void gfa_av(const float* __restrict__ qkv, const float* __restrict__ attn,
                       u16* __restrict__ out_tok) {
    __shared__ float at[1024];
    int bh = blockIdx.y; int b = bh >> 3, h = bh & 7;
    for (int i = threadIdx.x; i < 1024; i += 256) at[i] = attn[(size_t)bh * 1024 + i];
    __syncthreads();
    int n = blockIdx.x * 256 + threadIdx.x;
    if (n < Nt) {
        const float* vbase = qkv + ((size_t)(b * Nt + n)) * 768 + 512 + h * DH;
        float accd[32];
#pragma unroll
        for (int d = 0; d < 32; ++d) accd[d] = 0.0f;
        for (int e = 0; e < 32; ++e) {
            float ve = vbase[e];
#pragma unroll
            for (int d = 0; d < 32; ++d) accd[d] += at[d * 32 + e] * ve;
        }
        u16* ob = out_tok + ((size_t)(b * Nt + n)) * Cc + h * DH;
#pragma unroll
        for (int d = 0; d < 32; ++d) ob[d] = f2bf(accd[d]);
    }
}

// ---------- workspace layout (bytes, all 256-aligned) ----------
static constexpr size_t O_WQKV  = 0;                         // 768*256*2
static constexpr size_t O_WPROJ = 393216;                    // 256*256*2
static constexpr size_t O_WPW1  = 524288;                    // 1024*256*2
static constexpr size_t O_WPW2  = 1048576;                   // 256*1024*2
static constexpr size_t O_WCT   = 1572864;                   // 4*256*256*2
static constexpr size_t O_POOLX2= 2097152;                   // pooled f32 [B,C,784] -> later x2 f32 [M1,C]
static constexpr size_t O_TFP   = 14942208;                  // t f32 [M1,C]
static constexpr size_t O_TBF   = 27787264;                  // t bf16 -> later out_tok bf16 [M1,C]
static constexpr size_t O_QKVY  = 34209792;                  // qkv f32 [M1,768] -> later y NHWC f32 [B,56,56,C]
static constexpr size_t O_RQ    = 85590016;                  // [128,32] f32
static constexpr size_t O_RK    = 85606400;
static constexpr size_t O_ATTN  = 85622784;                  // [128,1024] f32
static constexpr size_t O_X2BF  = 86147072;                  // x2 bf16 [M1,C]
static constexpr size_t O_YBF   = 92569600;                  // y bf16 [M2,C]
static constexpr size_t O_GBF   = 118259712;                 // gelu bf16 [M2,4C]
// total 221,020,160 bytes

extern "C" void kernel_launch(void* const* d_in, const int* in_sizes, int n_in,
                              void* d_out, int out_size, void* d_ws, size_t ws_size,
                              hipStream_t stream) {
    (void)in_sizes; (void)n_in; (void)out_size; (void)ws_size;
    const float* x      = (const float*)d_in[0];
    const float* mask_u = (const float*)d_in[1];
    const float* dw_w   = (const float*)d_in[2];
    const float* dw_b   = (const float*)d_in[3];
    const float* g1     = (const float*)d_in[4];
    const float* b1     = (const float*)d_in[5];
    const float* qkv_w  = (const float*)d_in[6];
    const float* qkv_b  = (const float*)d_in[7];
    const float* temp   = (const float*)d_in[8];
    const float* proj_w = (const float*)d_in[9];
    const float* proj_b = (const float*)d_in[10];
    const float* ct_w   = (const float*)d_in[11];
    const float* ct_b   = (const float*)d_in[12];
    const float* g2     = (const float*)d_in[13];
    const float* b2     = (const float*)d_in[14];
    const float* pw1_w  = (const float*)d_in[15];
    const float* pw1_b  = (const float*)d_in[16];
    const float* pw2_w  = (const float*)d_in[17];
    const float* pw2_b  = (const float*)d_in[18];

    char* ws = (char*)d_ws;
    u16*   wqkv  = (u16*)(ws + O_WQKV);
    u16*   wproj = (u16*)(ws + O_WPROJ);
    u16*   wpw1  = (u16*)(ws + O_WPW1);
    u16*   wpw2  = (u16*)(ws + O_WPW2);
    u16*   wct   = (u16*)(ws + O_WCT);
    float* pooled= (float*)(ws + O_POOLX2);
    float* x2f   = (float*)(ws + O_POOLX2);   // alias (pooled dead after ln1)
    float* tf    = (float*)(ws + O_TFP);
    u16*   tb    = (u16*)(ws + O_TBF);
    u16*   outtok= (u16*)(ws + O_TBF);        // alias (t bf16 dead after qkv gemm)
    float* qkvf  = (float*)(ws + O_QKVY);
    float* ynhwc = (float*)(ws + O_QKVY);     // alias (qkv dead after av)
    float* rq    = (float*)(ws + O_RQ);
    float* rk    = (float*)(ws + O_RK);
    float* attn  = (float*)(ws + O_ATTN);
    u16*   x2bf  = (u16*)(ws + O_X2BF);
    u16*   ybf   = (u16*)(ws + O_YBF);
    u16*   gbf   = (u16*)(ws + O_GBF);
    float* out   = (float*)d_out;

    // weights -> bf16
    gfa_f2bf<<<(768 * 256 + 255) / 256, 256, 0, stream>>>(qkv_w, wqkv, 768 * 256);
    gfa_f2bf<<<(256 * 256 + 255) / 256, 256, 0, stream>>>(proj_w, wproj, 256 * 256);
    gfa_f2bf<<<(1024 * 256 + 255) / 256, 256, 0, stream>>>(pw1_w, wpw1, 1024 * 256);
    gfa_f2bf<<<(256 * 1024 + 255) / 256, 256, 0, stream>>>(pw2_w, wpw2, 256 * 1024);
    gfa_ctpack<<<(4 * 256 * 256 + 255) / 256, 256, 0, stream>>>(ct_w, wct);

    // dwconv + residual + pool, then LN1 -> tokens
    gfa_dwpool<<<Bb * Cc, 256, 0, stream>>>(x, dw_w, dw_b, pooled);
    gfa_ln1<<<M1, 256, 0, stream>>>(pooled, g1, b1, tf, tb);

    // qkv GEMM: [M1,256] x [768,256]^T -> f32 [M1,768]
    gfa_gemm_bf16<<<dim3(768 / 128, M1 / 64), 256, 0, stream>>>(
        tb, wqkv, qkv_b, nullptr, qkvf, nullptr, M1, 768, 256, 0, 0, 0);

    // attention (channel cross-covariance)
    gfa_norms<<<Bb * NH, 256, 0, stream>>>(qkvf, rq, rk);
    gfa_attn<<<Bb * NH, 256, 0, stream>>>(qkvf, rq, rk, mask_u, temp, attn);
    gfa_av<<<dim3((Nt + 255) / 256, Bb * NH), 256, 0, stream>>>(qkvf, attn, outtok);

    // proj GEMM + t residual -> x2 (f32 + bf16)
    gfa_gemm_bf16<<<dim3(256 / 128, M1 / 64), 256, 0, stream>>>(
        outtok, wproj, proj_b, tf, x2f, x2bf, M1, 256, 256, 0, 0, 0);

    // 2x2 stride-2 transposed conv = 4 GEMMs scattering into NHWC [B,56,56,C]
    for (int p = 0; p < 4; ++p) {
        gfa_gemm_bf16<<<dim3(256 / 128, M1 / 64), 256, 0, stream>>>(
            x2bf, wct + (size_t)p * 65536, ct_b, nullptr, ynhwc, nullptr,
            M1, 256, 256, 0, 1, p);
    }

    // LN2 -> bf16 tokens [M2,C]
    gfa_ln2<<<M2, 256, 0, stream>>>(ynhwc, g2, b2, ybf);

    // MLP: pw1 (+exact GELU) -> bf16 ; pw2 -> NCHW + inp0 residual -> d_out
    gfa_gemm_bf16<<<dim3(1024 / 128, M2 / 64), 256, 0, stream>>>(
        ybf, wpw1, pw1_b, nullptr, nullptr, gbf, M2, 1024, 256, 1, 0, 0);
    gfa_gemm_bf16<<<dim3(256 / 128, M2 / 64), 256, 0, stream>>>(
        gbf, wpw2, pw2_b, x, out, nullptr, M2, 256, 1024, 0, 2, 0);
}
